// GCN_51926154609285
// MI455X (gfx1250) — matrix operand order repaired and checked by reading, not compiled, not run on hardware
//
#include <hip/hip_runtime.h>
#include <hip/hip_bf16.h>
#include <stdint.h>

// ---------------------------------------------------------------------------
// Hetero-GraphSAGE on gfx1250: bf16 WMMA GEMMs + packed-bf16 atomic scatter.
// ---------------------------------------------------------------------------

typedef __attribute__((ext_vector_type(16))) __bf16 v16bf;
typedef __attribute__((ext_vector_type(8)))  float  v8f;

#define LDF 320   // padded feature stride (elements), multiple of 32
#define NGR 256   // graphs

// ------------------------------- utility -----------------------------------
__global__ void k_zero_f32(float* __restrict__ p, long n) {
  long i = (long)blockIdx.x * blockDim.x + threadIdx.x;
  if (i < n) p[i] = 0.0f;
}

// fp32 [rows x f] -> bf16 [rows x kpad] (zero padded)
__global__ void k_cvt_in(const float* __restrict__ x, __bf16* __restrict__ out,
                         long rows, int f, int kpad) {
  long i = (long)blockIdx.x * blockDim.x + threadIdx.x;
  if (i >= rows * (long)kpad) return;
  long r = i / kpad; int k = (int)(i % kpad);
  float v = (k < f) ? x[r * (long)f + k] : 0.0f;
  out[i] = (__bf16)v;
}

// W [k x 256] fp32 (+ optional W2) -> transposed Wt [256 x kpad] bf16
__global__ void k_wprep(const float* __restrict__ w1, const float* __restrict__ w2,
                        int k, int kpad, __bf16* __restrict__ wt) {
  int i = blockIdx.x * blockDim.x + threadIdx.x;
  if (i >= 256 * kpad) return;
  int n = i / kpad, kk = i % kpad;
  float v = 0.0f;
  if (kk < k) {
    v = w1[(long)kk * 256 + n];
    if (w2) v += w2[(long)kk * 256 + n];
  }
  wt[i] = (__bf16)v;
}

__global__ void k_bias_sum(const float* __restrict__ b1,
                           const float* __restrict__ b2, float* __restrict__ bs) {
  int i = threadIdx.x;
  float v = b1[i];
  if (b2) v += b2[i];
  bs[i] = v;
}

// ----------------------- scatter (segment-sum + count) ---------------------
// Packed bf16 atomics: source features are already bf16, so scatter without
// any conversion — 4x global_atomic_pk_add_bf16 per 8-feature chunk.
__device__ __forceinline__ void pk_add_bf16(unsigned long long addr, unsigned v0,
                                            unsigned v1, unsigned v2, unsigned v3) {
  asm volatile("global_atomic_pk_add_bf16 %0, %1, off\n\t"
               "global_atomic_pk_add_bf16 %0, %2, off offset:4\n\t"
               "global_atomic_pk_add_bf16 %0, %3, off offset:8\n\t"
               "global_atomic_pk_add_bf16 %0, %4, off offset:12"
               :: "v"(addr), "v"(v0), "v"(v1), "v"(v2), "v"(v3)
               : "memory");
}

__global__ void k_scatter(const int* __restrict__ ei, long ecnt,
                          const __bf16* __restrict__ feat, long ldf,
                          __bf16* __restrict__ acc, long lda,
                          float* __restrict__ cnt, int kchunks) {
  long t = (long)blockIdx.x * blockDim.x + threadIdx.x;
  if (t >= ecnt * (long)kchunks) return;
  long e = t / kchunks; int c = (int)(t % kchunks);
  int s = ei[e];
  int d = ei[ecnt + e];
  uint4 raw = *(const uint4*)(feat + (size_t)s * ldf + c * 8);   // 8 bf16
  unsigned long long ap = (unsigned long long)(acc + (size_t)d * lda + c * 8);
  pk_add_bf16(ap, raw.x, raw.y, raw.z, raw.w);
  if (c == 0) atomicAdd(cnt + d, 1.0f);
}

// --------------------------- WMMA fragment loads ----------------------------
// A (16x32 bf16): lane holds row lane&15; 16 contiguous K at k0+16*hi
__device__ __forceinline__ v16bf frag_a_bf16(const __bf16* p) {
  union { v16bf v; float4 q[2]; } u;
  u.q[0] = *(const float4*)(p);
  u.q[1] = *(const float4*)(p + 8);
  return u.v;
}
// B (32x16 bf16): lane holds col lane&15; 16 contiguous K at k0+16*hi (Wt pre-transposed)
__device__ __forceinline__ v16bf frag_b_bf16(const __bf16* p) {
  union { v16bf v; float4 q[2]; } u;
  u.q[0] = *(const float4*)(p);
  u.q[1] = *(const float4*)(p + 8);
  return u.v;
}

// --------------------------- fused hetero GEMM ------------------------------
// out = relu( A0@W0t^T + rowscale1*(A1@W1t^T) + rowscale2*(A2@W2t^T) + bias )
// All A sources bf16 (A1/A2 are packed-atomic neighbor sums; rowscale = 1/deg
// applied in the fp32 epilogue per accumulator set — no VALU in the K loops).
// block = 256 thr = 8 waves -> 32x256 tile; wave = 16x64 (4 wmma accs / source).
__global__ __launch_bounds__(256) void k_fused_gemm(
    const __bf16* __restrict__ a0, long lda0, const __bf16* __restrict__ w0t,
    const __bf16* __restrict__ a1, const float* __restrict__ cnt1, long lda1,
    const __bf16* __restrict__ w1t,
    const __bf16* __restrict__ a2, const float* __restrict__ cnt2, long lda2,
    const __bf16* __restrict__ w2t,
    const float* __restrict__ bias, __bf16* __restrict__ out, long ldo,
    int m, int k, int ldw) {
  const int lane  = threadIdx.x & 31;
  const int wave  = threadIdx.x >> 5;
  const int rbase = blockIdx.x * 32 + (wave & 1) * 16;
  const int cbase = (wave >> 1) * 64;
  const int nlane = lane & 15;
  const int hi    = lane >> 4;
  int row = rbase + nlane;
  if (row >= m) row = m - 1;   // clamp (EXEC must stay all-ones for WMMA)

  v8f vz = {0.f, 0.f, 0.f, 0.f, 0.f, 0.f, 0.f, 0.f};
  v8f acc0[4] = {vz, vz, vz, vz};
  v8f acc1[4] = {vz, vz, vz, vz};
  v8f acc2[4] = {vz, vz, vz, vz};

  {  // source 0: dst features @ summed Wr
    const __bf16* ap = a0 + (size_t)row * lda0 + hi * 16;
    for (int k0 = 0; k0 < k; k0 += 32) {
      v16bf a = frag_a_bf16(ap + k0);
#pragma unroll
      for (int j = 0; j < 4; j++) {
        v16bf b = frag_b_bf16(w0t + (size_t)(cbase + j * 16 + nlane) * ldw + hi * 16 + k0);
        acc0[j] = __builtin_amdgcn_wmma_f32_16x16x32_bf16(
            false, a, false, b, (short)0, acc0[j], false, false);
      }
    }
  }
  if (a1) {  // source 1: neighbor-sum @ Wl (scaled to mean in epilogue)
    const __bf16* ap = a1 + (size_t)row * lda1 + hi * 16;
    for (int k0 = 0; k0 < k; k0 += 32) {
      v16bf a = frag_a_bf16(ap + k0);
#pragma unroll
      for (int j = 0; j < 4; j++) {
        v16bf b = frag_b_bf16(w1t + (size_t)(cbase + j * 16 + nlane) * ldw + hi * 16 + k0);
        acc1[j] = __builtin_amdgcn_wmma_f32_16x16x32_bf16(
            false, a, false, b, (short)0, acc1[j], false, false);
      }
    }
  }
  if (a2) {  // source 2: second incoming edge type
    const __bf16* ap = a2 + (size_t)row * lda2 + hi * 16;
    for (int k0 = 0; k0 < k; k0 += 32) {
      v16bf a = frag_a_bf16(ap + k0);
#pragma unroll
      for (int j = 0; j < 4; j++) {
        v16bf b = frag_b_bf16(w2t + (size_t)(cbase + j * 16 + nlane) * ldw + hi * 16 + k0);
        acc2[j] = __builtin_amdgcn_wmma_f32_16x16x32_bf16(
            false, a, false, b, (short)0, acc2[j], false, false);
      }
    }
  }

  // epilogue: mean scaling + bias + relu + bf16 store
  // C layout: vgpr r -> row rbase+8*hi+r, col = cbase+16*j+(lane&15)
  float rs1[8], rs2[8];
#pragma unroll
  for (int r = 0; r < 8; r++) {
    int orow = rbase + hi * 8 + r;
    int cr = orow < m ? orow : m - 1;
    rs1[r] = cnt1 ? 1.0f / fmaxf(cnt1[cr], 1.0f) : 0.0f;
    rs2[r] = cnt2 ? 1.0f / fmaxf(cnt2[cr], 1.0f) : 0.0f;
  }
#pragma unroll
  for (int j = 0; j < 4; j++) {
    int col = cbase + j * 16 + nlane;
    float bv = bias[col];
#pragma unroll
    for (int r = 0; r < 8; r++) {
      int orow = rbase + hi * 8 + r;
      if (orow < m) {
        float v = acc0[j][r] + rs1[r] * acc1[j][r] + rs2[r] * acc2[j][r] + bv;
        out[(size_t)orow * ldo + col] = (__bf16)fmaxf(v, 0.0f);
      }
    }
  }
}

// -------------------------- pooling + classifier ----------------------------
__global__ void k_pool(const __bf16* __restrict__ feat, long ldf,
                       const int* __restrict__ batch, long rows,
                       float* __restrict__ pooled, float* __restrict__ pcnt,
                       int col_off) {
  long t = (long)blockIdx.x * blockDim.x + threadIdx.x;
  if (t >= rows * 32) return;
  long i = t >> 5; int c = (int)(t & 31);
  int g = batch[i];
  const __bf16* fp = feat + (size_t)i * ldf + c * 8;
  float4 raw = *(const float4*)fp;
  const __bf16* bp = (const __bf16*)&raw;
  float* dst = pooled + (size_t)g * 768 + col_off + c * 8;
#pragma unroll
  for (int j = 0; j < 8; j++) atomicAdd(dst + j, (float)bp[j]);
  if (c == 0) atomicAdd(pcnt + g, 1.0f);
}

__global__ void k_final(const float* __restrict__ pooled, const float* __restrict__ pcnt,
                        const float* __restrict__ w, const float* __restrict__ b,
                        float* __restrict__ out) {
  int t = blockIdx.x * blockDim.x + threadIdx.x;
  if (t >= NGR * 2) return;
  int g = t >> 1, c = t & 1;
  float acc = b[c];
  for (int tt = 0; tt < 3; tt++) {
    float rs = 1.0f / fmaxf(pcnt[tt * NGR + g], 1.0f);
    const float* pg = pooled + (size_t)g * 768 + tt * 256;
    for (int f = 0; f < 256; f++)
      acc += pg[f] * rs * w[(tt * 256 + f) * 2 + c];
  }
  out[t] = acc;
}

// ------------------------------ host driver ---------------------------------
extern "C" void kernel_launch(void* const* d_in, const int* in_sizes, int n_in,
                              void* d_out, int out_size, void* d_ws, size_t ws_size,
                              hipStream_t stream) {
  (void)n_in; (void)out_size; (void)ws_size;
  const long NA = 256, NT = 100000, NU = 200000;
  const long Nn[3] = {NA, NT, NU};

  const float* xin[3] = {(const float*)d_in[0], (const float*)d_in[1], (const float*)d_in[2]};
  const int* ei[5];
  long E[5];
  for (int e = 0; e < 5; e++) { ei[e] = (const int*)d_in[3 + e]; E[e] = in_sizes[3 + e] / 2; }
  const int* batch[3] = {(const int*)d_in[8], (const int*)d_in[9], (const int*)d_in[10]};

  // params: per layer, per edge {Wl, bl, Wr} — order disambiguated via in_sizes
  // edge order: 0=at 1=ta 2=tu 3=ut 4=uu
  const float *Wl[2][5], *Wr[2][5], *Bl[2][5];
  int p = 11;
  for (int l = 0; l < 2; l++)
    for (int e = 0; e < 5; e++) {
      Wl[l][e] = (const float*)d_in[p];
      if (in_sizes[p + 1] == 256) {  // insertion order: Wl, bl, Wr
        Bl[l][e] = (const float*)d_in[p + 1];
        Wr[l][e] = (const float*)d_in[p + 2];
      } else {                       // sorted order: Wl, Wr, bl
        Wr[l][e] = (const float*)d_in[p + 1];
        Bl[l][e] = (const float*)d_in[p + 2];
      }
      p += 3;
    }
  const float* lin_w = (const float*)d_in[p];
  const float* lin_b = (const float*)d_in[p + 1];

  // ---- workspace carve-out ----
  char* ws = (char*)d_ws;
  size_t off = 0;
  auto alloc = [&](size_t bytes) -> char* {
    char* r = ws + off;
    off += (bytes + 255) & ~(size_t)255;
    return r;
  };
  __bf16* cur[3], *nxt[3];
  for (int t = 0; t < 3; t++) cur[t] = (__bf16*)alloc((size_t)Nn[t] * LDF * 2);
  for (int t = 0; t < 3; t++) nxt[t] = (__bf16*)alloc((size_t)Nn[t] * LDF * 2);
  __bf16* accA = (__bf16*)alloc((size_t)NU * LDF * 2);   // packed-bf16 accumulators
  __bf16* accB = (__bf16*)alloc((size_t)NU * LDF * 2);
  float* cntA = (float*)alloc((size_t)NU * 4);
  float* cntB = (float*)alloc((size_t)NU * 4);
  __bf16* wlT[2][5];
  __bf16* wrT[2][3];
  float* bsum[2][3];
  for (int l = 0; l < 2; l++) {
    int kpad = l ? 256 : 320;
    for (int e = 0; e < 5; e++) wlT[l][e] = (__bf16*)alloc((size_t)256 * kpad * 2);
    for (int t = 0; t < 3; t++) wrT[l][t] = (__bf16*)alloc((size_t)256 * kpad * 2);
    for (int t = 0; t < 3; t++) bsum[l][t] = (float*)alloc(256 * 4);
  }
  float* pooled = (float*)alloc((size_t)NGR * 768 * 4);
  float* pcnt   = (float*)alloc((size_t)3 * NGR * 4);

  auto nb = [](long n, int b) { return (unsigned)((n + b - 1) / b); };

  // ---- preprocess: bf16 features + transposed bf16 weights + summed Wr/bias ----
  for (int t = 0; t < 3; t++)
    k_cvt_in<<<nb(Nn[t] * LDF, 256), 256, 0, stream>>>(xin[t], cur[t], Nn[t], 300, LDF);

  for (int l = 0; l < 2; l++) {
    int K = l ? 256 : 300, kpad = l ? 256 : 320;
    for (int e = 0; e < 5; e++)
      k_wprep<<<nb(256 * kpad, 256), 256, 0, stream>>>(Wl[l][e], nullptr, K, kpad, wlT[l][e]);
    // dst article: Wr_ta ; dst tweet: Wr_at+Wr_ut ; dst user: Wr_tu+Wr_uu
    k_wprep<<<nb(256 * kpad, 256), 256, 0, stream>>>(Wr[l][1], nullptr, K, kpad, wrT[l][0]);
    k_wprep<<<nb(256 * kpad, 256), 256, 0, stream>>>(Wr[l][0], Wr[l][3], K, kpad, wrT[l][1]);
    k_wprep<<<nb(256 * kpad, 256), 256, 0, stream>>>(Wr[l][2], Wr[l][4], K, kpad, wrT[l][2]);
    k_bias_sum<<<1, 256, 0, stream>>>(Bl[l][1], nullptr,  bsum[l][0]);
    k_bias_sum<<<1, 256, 0, stream>>>(Bl[l][0], Bl[l][3], bsum[l][1]);
    k_bias_sum<<<1, 256, 0, stream>>>(Bl[l][2], Bl[l][4], bsum[l][2]);
  }

  // edge src/dst node types, incoming edges per dst type
  const int src_of[5] = {0, 1, 1, 2, 2};
  const int inc[3][2] = {{1, -1}, {0, 3}, {2, 4}};

  // ---- two SAGE layers ----
  for (int l = 0; l < 2; l++) {
    int K = l ? 256 : 320;       // padded K (layer-1 pad region is all zeros)
    int kch = K / 8;
    for (int d = 0; d < 3; d++) {
      int e1 = inc[d][0], e2 = inc[d][1];
      k_zero_f32<<<nb(Nn[d] * LDF / 2, 256), 256, 0, stream>>>((float*)accA, Nn[d] * LDF / 2);
      k_zero_f32<<<nb(Nn[d], 256), 256, 0, stream>>>(cntA, Nn[d]);
      k_scatter<<<nb(E[e1] * kch, 256), 256, 0, stream>>>(
          ei[e1], E[e1], cur[src_of[e1]], LDF, accA, LDF, cntA, kch);
      if (e2 >= 0) {
        k_zero_f32<<<nb(Nn[d] * LDF / 2, 256), 256, 0, stream>>>((float*)accB, Nn[d] * LDF / 2);
        k_zero_f32<<<nb(Nn[d], 256), 256, 0, stream>>>(cntB, Nn[d]);
        k_scatter<<<nb(E[e2] * kch, 256), 256, 0, stream>>>(
            ei[e2], E[e2], cur[src_of[e2]], LDF, accB, LDF, cntB, kch);
      }
      k_fused_gemm<<<nb(Nn[d], 32), 256, 0, stream>>>(
          cur[d], LDF, wrT[l][d],
          accA, cntA, LDF, wlT[l][e1],
          e2 >= 0 ? accB : nullptr, e2 >= 0 ? cntB : nullptr, LDF,
          e2 >= 0 ? wlT[l][e2] : nullptr,
          bsum[l][d], nxt[d], LDF, (int)Nn[d], K, K);
    }
    for (int t = 0; t < 3; t++) { __bf16* tmp = cur[t]; cur[t] = nxt[t]; nxt[t] = tmp; }
  }

  // ---- global mean pool + linear classifier ----
  k_zero_f32<<<nb((long)NGR * 768, 256), 256, 0, stream>>>(pooled, (long)NGR * 768);
  k_zero_f32<<<nb(3 * NGR, 256), 256, 0, stream>>>(pcnt, 3 * NGR);
  for (int t = 0; t < 3; t++)
    k_pool<<<nb(Nn[t] * 32, 256), 256, 0, stream>>>(
        cur[t], LDF, batch[t], Nn[t], pooled, pcnt + t * NGR, t * 256);
  k_final<<<2, 256, 0, stream>>>(pooled, pcnt, lin_w, lin_b, (float*)d_out);
}
